// GATLayer_670014899155
// MI455X (gfx1250) — matrix operand order repaired and checked by reading, not compiled, not run on hardware
//
#include <hip/hip_runtime.h>

#define B_     8
#define N_     2048
#define DIN    128
#define DOUT   64
#define ALPHA  0.2f
#define CHUNKS 64          // N_/32
#define PITCH  34          // padded LDS row pitch (halfs) -> bank-conflict free transpose

typedef __attribute__((ext_vector_type(16))) _Float16 v16h;
typedef __attribute__((ext_vector_type(8)))  float    v8f;

__device__ __forceinline__ float lrelu(float v) { return v > 0.f ? v : ALPHA * v; }

// ---------------------------------------------------------------------------
// Kernel 1: xp = x @ W^T + b  (f32), emit:
//   - xpswz : xp cast to f16, stored pre-swizzled in WMMA-B (32x16 f16) layout
//   - es/ed : e_src[m] = xp[m]·a_src, e_dst[n] = xp[n]·a_dst  (f32)
// One block per (batch,node); 64 threads = one output feature each.
// ---------------------------------------------------------------------------
__global__ __launch_bounds__(64) void gat_proj_kernel(
    const float* __restrict__ x, const float* __restrict__ W,
    const float* __restrict__ bias, const float* __restrict__ a,
    _Float16* __restrict__ xpswz, float* __restrict__ es, float* __restrict__ ed)
{
  __shared__ float xs[DIN];
  __shared__ float red[2 * DOUT];
  const int bi = blockIdx.x / N_;
  const int n  = blockIdx.x % N_;
  const int t  = threadIdx.x;                    // output feature d
  const float* xrow = x + (size_t)blockIdx.x * DIN;
  xs[t]      = xrow[t];
  xs[t + 64] = xrow[t + 64];
  __syncthreads();

  float acc = bias[t];
  const float* wr = W + t * DIN;
#pragma unroll 8
  for (int k = 0; k < DIN; ++k) acc = fmaf(xs[k], wr[k], acc);

  // swizzled f16 store into WMMA-B layout:
  //   chunk c = n/32, ko = n%32 ; B element K=ko lives at lane = 16*(ko>>4) + (d%16),
  //   half index h = ko&15, d-block = d/16.
  const int c    = n >> 5;
  const int ko   = n & 31;
  const int hiB  = ko >> 4;
  const int hB   = ko & 15;
  const int dblk = t >> 4;
  const int lane = (hiB << 4) | (t & 15);
  const size_t off = ((((size_t)(bi * CHUNKS + c)) * 4 + dblk) * 32 + lane) * 16 + hB;
  xpswz[off] = (_Float16)acc;

  red[t]        = acc * a[t];                    // a_src = a[0][0:64]
  red[DOUT + t] = acc * a[DOUT + t];             // a_dst = a[0][64:128]
  __syncthreads();
  if (t < 2) {
    float s = 0.f;
#pragma unroll
    for (int k = 0; k < DOUT; ++k) s += red[t * DOUT + k];
    if (t == 0) es[blockIdx.x] = s;
    else        ed[blockIdx.x] = s;
  }
}

// ---------------------------------------------------------------------------
// Kernel 2: fused masked-softmax attention + WMMA (att @ xp).
// One workgroup (4 waves / 128 thr) handles a 16-row m-tile of one batch.
// Pass 1 (RT loads, fills L2): masked max of d[n] per row; lrelu monotone =>
//   rowmax = lrelu(s + dmax).
// Pass 2 (NT loads, last use of adj): p~ = exp(lrelu(s+d)-rowmax) masked;
//   rowsum accumulated alongside; p~ tile transposed via padded LDS into the
//   WMMA-A layout; B loaded as one contiguous v16h per lane from the
//   pre-swizzled xp buffer; normalization deferred to the epilogue.
// ---------------------------------------------------------------------------
__global__ __launch_bounds__(128) void gat_attn_kernel(
    const int* __restrict__ adj, const _Float16* __restrict__ xpswz,
    const float* __restrict__ es, const float* __restrict__ ed,
    float* __restrict__ out)
{
  __shared__ float     s_sh[16];
  __shared__ float     dmax_sh[4 * 16];
  __shared__ float     sum_sh[4 * 16];
  __shared__ _Float16  att_sh[4 * 16 * PITCH];     // per-wave transpose staging
  __shared__ float     c_sh[4][4][8 * 32];         // [wave][dblk][r*32+lane]

  const int bi     = blockIdx.x >> 7;              // 128 tiles per batch
  const int m0     = (blockIdx.x & 127) << 4;
  const int tid    = threadIdx.x;
  const int w      = tid >> 5;
  const int lane   = tid & 31;
  const int laneHi = lane >> 4;
  const int ln     = lane & 15;

  if (tid < 16) s_sh[tid] = es[bi * N_ + m0 + tid];

  const int*   __restrict__ adjB = adj + ((size_t)bi * N_ + m0) * N_;
  const float* __restrict__ edB  = ed + bi * N_;

  // -------- pass 1: masked per-row max of d (RT: prime L2 for pass 2) --------
  float dmx[16];
#pragma unroll
  for (int r = 0; r < 16; ++r) dmx[r] = -3.0e38f;
  for (int c = w; c < CHUNKS; c += 4) {
    const int n = (c << 5) + lane;
    const float dv = edB[n];
#pragma unroll
    for (int r = 0; r < 16; ++r) {
      const int av = adjB[(size_t)r * N_ + n];     // coalesced 128B row segment
      dmx[r] = fmaxf(dmx[r], av ? dv : -3.0e38f);
    }
  }
#pragma unroll
  for (int r = 0; r < 16; ++r) {
#pragma unroll
    for (int off = 16; off >= 1; off >>= 1)
      dmx[r] = fmaxf(dmx[r], __shfl_xor(dmx[r], off, 32));
  }
  if (lane == 0) {
#pragma unroll
    for (int r = 0; r < 16; ++r) dmax_sh[w * 16 + r] = dmx[r];
  }
  __syncthreads();

  float sreg[16], rmax[16];
#pragma unroll
  for (int r = 0; r < 16; ++r) {
    sreg[r] = s_sh[r];
    const float dm = fmaxf(fmaxf(dmax_sh[r], dmax_sh[16 + r]),
                           fmaxf(dmax_sh[32 + r], dmax_sh[48 + r]));
    rmax[r] = lrelu(sreg[r] + dm);                 // exact rowmax (diag guarantees finite)
  }

  // -------- pass 2: exp + WMMA accumulate (NT: last use of adj) --------
  const v8f vzero = {0.f, 0.f, 0.f, 0.f, 0.f, 0.f, 0.f, 0.f};
  v8f acc[4];
#pragma unroll
  for (int dblk = 0; dblk < 4; ++dblk) acc[dblk] = vzero;
  float sm[16];
#pragma unroll
  for (int r = 0; r < 16; ++r) sm[r] = 0.f;

  _Float16* mya = att_sh + w * (16 * PITCH);

  for (int c = w; c < CHUNKS; c += 4) {
    const int n = (c << 5) + lane;
    const float dv = edB[n];
#pragma unroll
    for (int r = 0; r < 16; ++r) {
      const int av = __builtin_nontemporal_load(&adjB[(size_t)r * N_ + n]);
      const float e = lrelu(sreg[r] + dv);
      const float p = av ? __expf(e - rmax[r]) : 0.f;
      sm[r] += p;
      mya[r * PITCH + lane] = (_Float16)p;         // lane = column n
    }
    __builtin_amdgcn_wave_barrier();               // keep LDS st->ld program order (HW in-order per wave)

    // Gather A in WMMA 16x32 f16 layout: lane M = ln, half h -> K
    v16h A;
#pragma unroll
    for (int h = 0; h < 16; ++h) {
      const int K = h + (laneHi << 4);
      A[h] = mya[ln * PITCH + K];
    }
    __builtin_amdgcn_wave_barrier();

    // B: one contiguous 32B v16h per lane per d-block (pre-swizzled, L2 resident)
    const v16h* __restrict__ Bp =
        (const v16h*)xpswz + ((size_t)(bi * CHUNKS + c) * 4) * 32 + lane;
#pragma unroll
    for (int dblk = 0; dblk < 4; ++dblk) {
      const v16h Bm = Bp[dblk * 32];
      acc[dblk] = __builtin_amdgcn_wmma_f32_16x16x32_f16(
          false, A, false, Bm, (short)0, acc[dblk], false, false);
    }
  }

  // -------- reductions across lanes/waves, normalize, store --------
#pragma unroll
  for (int r = 0; r < 16; ++r) {
#pragma unroll
    for (int off = 16; off >= 1; off >>= 1) sm[r] += __shfl_xor(sm[r], off, 32);
  }
  if (lane == 0) {
#pragma unroll
    for (int r = 0; r < 16; ++r) sum_sh[w * 16 + r] = sm[r];
  }
#pragma unroll
  for (int dblk = 0; dblk < 4; ++dblk) {
#pragma unroll
    for (int r = 0; r < 8; ++r) c_sh[w][dblk][r * 32 + lane] = acc[dblk][r];
  }
  __syncthreads();

  // wave w finalizes d-block w; C layout: element r at lane -> row = r + 8*laneHi, col = ln
#pragma unroll
  for (int r = 0; r < 8; ++r) {
    const int row = r + (laneHi << 3);
    const float v = c_sh[0][w][r * 32 + lane] + c_sh[1][w][r * 32 + lane]
                  + c_sh[2][w][r * 32 + lane] + c_sh[3][w][r * 32 + lane];
    const float tot = sum_sh[row] + sum_sh[16 + row]
                    + sum_sh[32 + row] + sum_sh[48 + row];
    __builtin_nontemporal_store(
        v / tot, &out[((size_t)(bi * N_ + m0 + row)) * DOUT + (w << 4) + ln]);
  }
}

// ---------------------------------------------------------------------------
extern "C" void kernel_launch(void* const* d_in, const int* in_sizes, int n_in,
                              void* d_out, int out_size, void* d_ws, size_t ws_size,
                              hipStream_t stream) {
  const float* x    = (const float*)d_in[0];
  const int*   adj  = (const int*)d_in[1];
  const float* W    = (const float*)d_in[2];
  const float* bias = (const float*)d_in[3];
  const float* a    = (const float*)d_in[4];
  float* out = (float*)d_out;

  char* ws = (char*)d_ws;
  _Float16* xpswz = (_Float16*)ws;                               // 2 MB
  float* es = (float*)(ws + (size_t)B_ * N_ * DOUT * sizeof(_Float16));
  float* ed = es + B_ * N_;

  hipLaunchKernelGGL(gat_proj_kernel, dim3(B_ * N_), dim3(64), 0, stream,
                     x, W, bias, a, xpswz, es, ed);
  hipLaunchKernelGGL(gat_attn_kernel, dim3(B_ * (N_ / 16)), dim3(128), 0, stream,
                     adj, xpswz, es, ed, out);
}
